// GRUSoftQNetwork_53463752901210
// MI455X (gfx1250) — matrix operand order repaired and checked by reading, not compiled, not run on hardware
//
#include <hip/hip_runtime.h>

// ---------------------------------------------------------------------------
// GRU Soft-Q network forward for MI455X (gfx1250, wave32, WMMA bf16).
// T=512, B=1024, S=64, A=10, H=128.
// ---------------------------------------------------------------------------

typedef __attribute__((ext_vector_type(16))) __bf16 bf16x16;
typedef __attribute__((ext_vector_type(8)))  __bf16 bf16x8;
typedef __attribute__((ext_vector_type(8)))  float  f32x8;

#define T_DIM 512
#define B_DIM 1024
#define S_D   64
#define A_D   10
#define H_D   128
#define TB    (T_DIM * B_DIM)

// D = A(16x32 bf16) * B(32x16 bf16) + C(16x16 f32)
__device__ __forceinline__ f32x8 wmma_bf16(bf16x16 a, bf16x16 b, f32x8 c) {
  return __builtin_amdgcn_wmma_f32_16x16x32_bf16(false, a, false, b, (short)0, c,
                                                 false, false);
}

// A-fragment (16x32, bf16) from a row-major bf16 buffer, row stride = kstride.
__device__ __forceinline__ bf16x16 load_afrag(const __bf16* buf, int kstride,
                                              int ktile, int lane) {
  int row  = lane & 15;
  int half = lane >> 4;
  const __bf16* p = buf + row * kstride + ktile * 32 + half * 16;
  return *(const bf16x16*)(p);
}

// B-fragment (32x16, bf16) for out = in @ W.T with W row-major [N][K].
__device__ __forceinline__ bf16x16 load_bfrag(const __bf16* W, int kstride,
                                              int ntile, int ktile, int lane) {
  int col  = lane & 15;
  int half = lane >> 4;
  const __bf16* p = W + (ntile * 16 + col) * kstride + ktile * 32 + half * 16;
  return *(const bf16x16*)(p);
}

// Async-copy 6x16B per lane of a contiguous 24KB tile: global -> LDS.
// ISA: INST_OFFSET is added to BOTH the LDS address (VDST vgpr) and the
// global address (SADDR + VADDR + offset), so one vgpr pair covers all 6.
__device__ __forceinline__ void issue_xg_copy(const float* gbase,
                                              unsigned lds_addr, unsigned vofs) {
  asm volatile(
      "global_load_async_to_lds_b128 %0, %1, %2 offset:0\n\t"
      "global_load_async_to_lds_b128 %0, %1, %2 offset:512\n\t"
      "global_load_async_to_lds_b128 %0, %1, %2 offset:1024\n\t"
      "global_load_async_to_lds_b128 %0, %1, %2 offset:1536\n\t"
      "global_load_async_to_lds_b128 %0, %1, %2 offset:2048\n\t"
      "global_load_async_to_lds_b128 %0, %1, %2 offset:2560"
      :: "v"(lds_addr), "v"(vofs), "s"(gbase) : "memory");
}

// ---------------------------------------------------------------------------
// Kernel 0: one-shot weight conversion f32 -> bf16 (L2-resident afterwards).
// ---------------------------------------------------------------------------
__global__ void prep_weights(const float* W1, const float* W2, const float* Wih,
                             const float* Whh, const float* Wc,
                             __bf16* W1b, __bf16* W2b, __bf16* Wihb,
                             __bf16* Whhb, __bf16* Wcb) {
  int i = blockIdx.x * blockDim.x + threadIdx.x;   // 0 .. 49151
  if (i < 8192)  W1b[i]  = (__bf16)W1[i];
  if (i < 16384) W2b[i]  = (__bf16)W2[i];
  if (i < 49152) { Wihb[i] = (__bf16)Wih[i]; Whhb[i] = (__bf16)Whh[i]; }
  if (i < 2048)  Wcb[i]  = (i < A_D * H_D) ? (__bf16)Wc[i] : (__bf16)0.0f;
}

// ---------------------------------------------------------------------------
// Kernel 1: encoder + input-gate projection, one 16-row tile per wave.
// ---------------------------------------------------------------------------
__global__ void encode_kernel(const float* __restrict__ x,
                              const float* __restrict__ b1,
                              const float* __restrict__ b2,
                              const float* __restrict__ b_ih,
                              const __bf16* __restrict__ W1b,
                              const __bf16* __restrict__ W2b,
                              const __bf16* __restrict__ Wihb,
                              float* __restrict__ xg) {
  __shared__ __bf16 stage_all[8][16 * H_D];        // 32 KB, one slab per wave
  const int wave = threadIdx.x >> 5;
  const int lane = threadIdx.x & 31;
  const int m0   = (blockIdx.x * 8 + wave) * 16;
  __bf16* st = stage_all[wave];

  const int colL  = lane & 15;
  const int rbase = (lane >> 4) * 8;

  // ---- stage x tile (16 x 64) as bf16 ----
  for (int idx = lane; idx < 16 * S_D; idx += 32) {
    int r = idx >> 6, c = idx & 63;
    st[r * S_D + c] = (__bf16)x[(m0 + r) * S_D + c];
  }

  // ---- GEMM1: (16x64) @ (64x128) ----
  bf16x16 ax0 = load_afrag(st, S_D, 0, lane);
  bf16x16 ax1 = load_afrag(st, S_D, 1, lane);
  f32x8 acc[8];
#pragma unroll
  for (int nt = 0; nt < 8; ++nt) {
    f32x8 c = {};
    c = wmma_bf16(ax0, load_bfrag(W1b, S_D, nt, 0, lane), c);
    c = wmma_bf16(ax1, load_bfrag(W1b, S_D, nt, 1, lane), c);
    acc[nt] = c;
  }
#pragma unroll
  for (int nt = 0; nt < 8; ++nt) {
    float bv = b1[nt * 16 + colL];
#pragma unroll
    for (int i = 0; i < 8; ++i) {
      float v = acc[nt][i] + bv;
      v = v > 0.0f ? v : 0.0f;
      st[(rbase + i) * H_D + nt * 16 + colL] = (__bf16)v;
    }
  }

  // ---- GEMM2: (16x128) @ (128x128) ----
  bf16x16 ah[4];
#pragma unroll
  for (int kt = 0; kt < 4; ++kt) ah[kt] = load_afrag(st, H_D, kt, lane);
#pragma unroll
  for (int nt = 0; nt < 8; ++nt) {
    f32x8 c = {};
#pragma unroll
    for (int kt = 0; kt < 4; ++kt)
      c = wmma_bf16(ah[kt], load_bfrag(W2b, H_D, nt, kt, lane), c);
    acc[nt] = c;
  }
#pragma unroll
  for (int nt = 0; nt < 8; ++nt) {
    float bv = b2[nt * 16 + colL];
#pragma unroll
    for (int i = 0; i < 8; ++i) {
      float v = acc[nt][i] + bv;
      v = v > 0.0f ? v : 0.0f;
      st[(rbase + i) * H_D + nt * 16 + colL] = (__bf16)v;
    }
  }

  // ---- GEMM3: (16x128) @ (128x384) -> xg ----
#pragma unroll
  for (int kt = 0; kt < 4; ++kt) ah[kt] = load_afrag(st, H_D, kt, lane);
  for (int nt = 0; nt < 24; ++nt) {
    f32x8 c = {};
#pragma unroll
    for (int kt = 0; kt < 4; ++kt)
      c = wmma_bf16(ah[kt], load_bfrag(Wihb, H_D, nt, kt, lane), c);
    float bv = b_ih[nt * 16 + colL];
#pragma unroll
    for (int i = 0; i < 8; ++i)
      xg[(size_t)(m0 + rbase + i) * 384 + nt * 16 + colL] = c[i] + bv;
  }
}

// ---------------------------------------------------------------------------
// Kernel 2: GRU scan fused with q = h @ Wc.T + bc.
// One workgroup per 16 batch rows; 8 waves each own 48 of 384 gate columns (16 per gate).
// W_hh B-fragments live in VGPRs all 512 steps; h lives in LDS; the 24KB
// xg tile for step t+1 is double-buffered into LDS with async copies.
// ---------------------------------------------------------------------------
__global__ void scan_kernel(const float* __restrict__ done,
                            const float* __restrict__ h0,
                            const float* __restrict__ b_hh,
                            const float* __restrict__ bc,
                            const __bf16* __restrict__ Whhb,
                            const __bf16* __restrict__ Wcb,
                            const float* __restrict__ xg,
                            float* __restrict__ q_out,
                            float* __restrict__ h_out) {
  __shared__ float  hf[16 * H_D];       // fp32 master copy of h tile (8KB)
  __shared__ __bf16 hb[16 * H_D];       // bf16 staging for A-fragments (4KB)
  __shared__ float  xs[2][16 * 384];    // xg double buffer (2 x 24KB)

  const int wave  = threadIdx.x >> 5;
  const int lane  = threadIdx.x & 31;
  const int bt    = blockIdx.x;
  const int col   = wave * 16 + (lane & 15);
  const int rbase = (lane >> 4) * 8;

  // Per-lane async-copy addressing: lane covers 6x16B chunks at +j*512B.
  const unsigned cofs = (unsigned)((wave * 192 + lane) * 16);
  const unsigned lds_x0 = (unsigned)(uintptr_t)(&xs[0][0]) + cofs;
  const unsigned lds_x1 = (unsigned)(uintptr_t)(&xs[1][0]) + cofs;

  // Preload recurrent weight B-fragments.
  bf16x16 wb[3][4];
#pragma unroll
  for (int g = 0; g < 3; ++g)
#pragma unroll
    for (int kt = 0; kt < 4; ++kt)
      wb[g][kt] = load_bfrag(Whhb, H_D, g * 8 + wave, kt, lane);
  float bh[3];
#pragma unroll
  for (int g = 0; g < 3; ++g) bh[g] = b_hh[g * H_D + col];

  bf16x16 wcf[4];
  float bcv = 0.0f;
  if (wave == 0) {
#pragma unroll
    for (int kt = 0; kt < 4; ++kt) wcf[kt] = load_bfrag(Wcb, H_D, 0, kt, lane);
    bcv = ((lane & 15) < A_D) ? bc[lane & 15] : 0.0f;
  }

  // init h from gru_state; kick off async copy of the t=0 xg tile
  issue_xg_copy(xg + (size_t)(bt * 16) * 384, lds_x0, cofs);
#pragma unroll
  for (int j = 0; j < 8; ++j) {
    int r = rbase + j;
    hf[r * H_D + col] = h0[(bt * 16 + r) * H_D + col];
  }
  __syncthreads();

  for (int t = 0; t < T_DIM; ++t) {
    const int rowg = t * B_DIM + bt * 16;

    // issue async copy for step t+1 into the other buffer
    if (t + 1 < T_DIM) {
      issue_xg_copy(xg + (size_t)(rowg + B_DIM) * 384,
                    ((t + 1) & 1) ? lds_x1 : lds_x0, cofs);
    }

    // phase 1: episode-reset mask; publish masked h as bf16
#pragma unroll
    for (int j = 0; j < 8; ++j) {
      int r = rbase + j;
      float d  = done[rowg + r];
      float hm = (1.0f - d) * hf[r * H_D + col];
      hf[r * H_D + col] = hm;
      hb[r * H_D + col] = (__bf16)hm;
    }
    __syncthreads();

    // phase 2: hg(:, my 16 cols of each gate) = h @ W_hh.T
    bf16x16 ahr[4];
#pragma unroll
    for (int kt = 0; kt < 4; ++kt) ahr[kt] = load_afrag(hb, H_D, kt, lane);
    f32x8 hg[3];
#pragma unroll
    for (int g = 0; g < 3; ++g) {
      f32x8 c = {};
#pragma unroll
      for (int kt = 0; kt < 4; ++kt) c = wmma_bf16(ahr[kt], wb[g][kt], c);
      hg[g] = c;
    }

    // drain this wave's older async batch (tile t); barrier makes the whole
    // 24KB tile visible to all waves before phase 3 consumes it.
    if (t + 1 < T_DIM) {
      asm volatile("s_wait_asynccnt 0x6" ::: "memory");
    } else {
      asm volatile("s_wait_asynccnt 0x0" ::: "memory");
    }
    __syncthreads();

    // phase 3: gates + state update (fp32), publish new h
    const float* xt = xs[t & 1];
#pragma unroll
    for (int j = 0; j < 8; ++j) {
      int r = rbase + j;
      float xr = xt[r * 384 + col];
      float xz = xt[r * 384 + H_D + col];
      float xn = xt[r * 384 + 2 * H_D + col];
      float rg = 1.0f / (1.0f + expf(-(xr + hg[0][j] + bh[0])));
      float zg = 1.0f / (1.0f + expf(-(xz + hg[1][j] + bh[1])));
      float ng = tanhf(xn + rg * (hg[2][j] + bh[2]));
      float hold = hf[r * H_D + col];
      float hnew = (1.0f - zg) * ng + zg * hold;
      hf[r * H_D + col] = hnew;
      hb[r * H_D + col] = (__bf16)hnew;
    }
    __syncthreads();

    // phase 4: q(:,0..9) = h_new @ Wc.T + bc   (wave 0, Wc zero-padded)
    if (wave == 0) {
      bf16x16 ahq[4];
#pragma unroll
      for (int kt = 0; kt < 4; ++kt) ahq[kt] = load_afrag(hb, H_D, kt, lane);
      f32x8 c = {};
#pragma unroll
      for (int kt = 0; kt < 4; ++kt) c = wmma_bf16(ahq[kt], wcf[kt], c);
      int a = lane & 15;
      if (a < A_D) {
#pragma unroll
        for (int j = 0; j < 8; ++j)
          q_out[(size_t)(rowg + rbase + j) * A_D + a] = c[j] + bcv;
      }
    }
    __syncthreads();
  }

  // final hidden state
#pragma unroll
  for (int j = 0; j < 8; ++j) {
    int r = rbase + j;
    h_out[(bt * 16 + r) * H_D + col] = hf[r * H_D + col];
  }
}

// ---------------------------------------------------------------------------
extern "C" void kernel_launch(void* const* d_in, const int* in_sizes, int n_in,
                              void* d_out, int out_size, void* d_ws, size_t ws_size,
                              hipStream_t stream) {
  const float* x    = (const float*)d_in[0];
  const float* done = (const float*)d_in[1];
  const float* h0   = (const float*)d_in[2];
  const float* W1   = (const float*)d_in[3];
  const float* b1   = (const float*)d_in[4];
  const float* W2   = (const float*)d_in[5];
  const float* b2   = (const float*)d_in[6];
  const float* W_ih = (const float*)d_in[7];
  const float* b_ih = (const float*)d_in[8];
  const float* W_hh = (const float*)d_in[9];
  const float* b_hh = (const float*)d_in[10];
  const float* Wc   = (const float*)d_in[11];
  const float* bc   = (const float*)d_in[12];

  char* ws = (char*)d_ws;
  __bf16* W1b  = (__bf16*)(ws + 0);        // 128*64
  __bf16* W2b  = (__bf16*)(ws + 16384);    // 128*128
  __bf16* Wihb = (__bf16*)(ws + 49152);    // 384*128
  __bf16* Whhb = (__bf16*)(ws + 147456);   // 384*128
  __bf16* Wcb  = (__bf16*)(ws + 245760);   // 16*128 (zero-padded)
  float*  xg   = (float*)(ws + 262144);    // T*B*384 f32

  float* q_out = (float*)d_out;                       // (T,B,10)
  float* h_out = q_out + (size_t)TB * A_D;            // (1,B,128)

  hipLaunchKernelGGL(prep_weights, dim3(192), dim3(256), 0, stream,
                     W1, W2, W_ih, W_hh, Wc, W1b, W2b, Wihb, Whhb, Wcb);

  hipLaunchKernelGGL(encode_kernel, dim3(TB / 16 / 8), dim3(256), 0, stream,
                     x, b1, b2, b_ih, W1b, W2b, Wihb, xg);

  hipLaunchKernelGGL(scan_kernel, dim3(B_DIM / 16), dim3(256), 0, stream,
                     done, h0, b_hh, bc, Whhb, Wcb, xg, q_out, h_out);
}
